// LlamaAttention_63290638074351
// MI455X (gfx1250) — matrix operand order repaired and checked by reading, not compiled, not run on hardware
//
#include <hip/hip_runtime.h>
#include <hip/hip_bf16.h>

// ---------------------------------------------------------------------------
// Llama attention block for MI455X (gfx1250), bf16 WMMA path.
//   B=2, S=2048, HID=4096, H=32, KV=8, D=128, G=4
// Stages:
//   0) fp32 -> bf16 conversion of hidden, w_qkv, w_o
//   1) QKV GEMM: LDS double-buffered via GLOBAL_LOAD_ASYNC_TO_LDS_B128 + WMMA
//   2) RoPE + scatter into Q[b,h,s,d], K[b,kv,s,d], V^T[b,kv,d,s]
//   3) Flash attention, one wave32 per 16-row Q tile (v_wmma_f32_16x16x32_bf16)
//   4) Output GEMM (same async-LDS WMMA kernel) -> fp32 d_out
// ---------------------------------------------------------------------------

typedef __bf16 bf16;
typedef __attribute__((ext_vector_type(16))) __bf16 v16bf;
typedef __attribute__((ext_vector_type(8)))  __bf16 v8bf;
typedef __attribute__((ext_vector_type(8)))  float  v8f;

#define CB 2
#define CS 2048
#define CHID 4096
#define CH 32
#define CKV 8
#define CD 128
#define CG 4
#define CQKV ((CH + 2 * CKV) * CD)   // 6144

__device__ __forceinline__ v16bf ldfrag(const bf16* p0, const bf16* p1) {
  union { v16bf f; v8bf h[2]; } u;
  u.h[0] = *reinterpret_cast<const v8bf*>(p0);
  u.h[1] = *reinterpret_cast<const v8bf*>(p1);
  return u.f;
}

__device__ __forceinline__ v8f wmma_bf16(v16bf a, v16bf b, v8f c) {
  return __builtin_amdgcn_wmma_f32_16x16x32_bf16(
      /*neg_a=*/false, a, /*neg_b=*/false, b,
      /*c_mod=*/(short)0, c, /*reuse_a=*/false, /*reuse_b=*/false);
}

// CDNA5 async copy: global -> LDS, 16B per lane, tracked by ASYNCcnt.
__device__ __forceinline__ void async_g2l_b128(unsigned lds_addr, const bf16* gptr) {
  asm volatile("global_load_async_to_lds_b128 %0, %1, off"
               :: "v"(lds_addr), "v"(gptr) : "memory");
}
__device__ __forceinline__ void wait_asynccnt0() {
  asm volatile("s_wait_asynccnt 0x0" ::: "memory");
}

// ---------------------------------------------------------------------------
// Stage 0: fp32 -> bf16
// ---------------------------------------------------------------------------
__global__ void cvt_f32_bf16(const float* __restrict__ src, bf16* __restrict__ dst, int n) {
  int i = blockIdx.x * blockDim.x + threadIdx.x;
  int stride = gridDim.x * blockDim.x;
  for (; i < n; i += stride) dst[i] = (bf16)src[i];
}

// ---------------------------------------------------------------------------
// Stages 1 & 4: C[M,N] = A[M,K] @ Bw[N,K]^T   (NT GEMM, bf16 in, f32 acc)
// 256-thread block computes a 128x128 C macro-tile; 8 waves in a 4(M)x2(N)
// grid, each wave a 32x64 tile (2x4 accumulators). A/B panels (K-depth 64)
// are double-buffered in LDS via async global->LDS copies. Fragments are
// preloaded per K-chunk so DS loads clause-up and overlap the WMMA burst.
// ---------------------------------------------------------------------------
__global__ void __launch_bounds__(256)
wmma_gemm_nt_lds(const bf16* __restrict__ A, const bf16* __restrict__ Bw,
                 float* __restrict__ Cf, bf16* __restrict__ Cb,
                 int M, int N, int K) {
  __shared__ bf16 Asmem[2][128 * 64];   // 2 x 16 KB
  __shared__ bf16 Bsmem[2][128 * 64];   // 2 x 16 KB

  const int tid  = threadIdx.x;
  const int lane = tid & 31;
  const int wv   = tid >> 5;            // wave 0..7
  const int wm   = wv & 3;              // M sub-tile (32 rows each)
  const int wn   = wv >> 2;             // N sub-tile (64 cols each)
  const int r    = lane & 15;
  const int kh   = lane >> 4;

  const int ntiles = N >> 7;            // 128-wide macro tiles along N
  const int mBase  = (blockIdx.x / ntiles) * 128;
  const int nBase  = (blockIdx.x % ntiles) * 128;

  v8f acc[2][4];
  #pragma unroll
  for (int mi = 0; mi < 2; ++mi)
    #pragma unroll
    for (int t = 0; t < 4; ++t) acc[mi][t] = (v8f)(0.0f);

  // ---- async stage issue: A 128x64 + B 128x64, 4+4 x 16B per thread ----
  auto issue_stage = [&](int k0, int buf) {
    #pragma unroll
    for (int it = 0; it < 4; ++it) {
      int seg = tid + it * 256;          // 1024 x 16B segments
      int row = seg >> 3, c8 = seg & 7;
      const bf16* g = A + (size_t)(mBase + row) * K + k0 + c8 * 8;
      unsigned l = (unsigned)(size_t)&Asmem[buf][row * 64 + c8 * 8];
      async_g2l_b128(l, g);
    }
    #pragma unroll
    for (int it = 0; it < 4; ++it) {
      int seg = tid + it * 256;          // 1024 x 16B segments
      int row = seg >> 3, c8 = seg & 7;
      const bf16* g = Bw + (size_t)(nBase + row) * K + k0 + c8 * 8;
      unsigned l = (unsigned)(size_t)&Bsmem[buf][row * 64 + c8 * 8];
      async_g2l_b128(l, g);
    }
  };

  // ---- compute one staged K-depth-64 panel: 2 chunks x 8 WMMAs/wave ------
  auto compute_stage = [&](int buf) {
    #pragma unroll
    for (int kc = 0; kc < 2; ++kc) {
      // preload all fragments of this chunk, then burst the WMMAs
      const bf16* ap0 = &Asmem[buf][(wm * 32 + r) * 64 + kc * 32];
      const bf16* ap1 = &Asmem[buf][(wm * 32 + 16 + r) * 64 + kc * 32];
      v16bf a0 = ldfrag(ap0 + kh * 16, ap0 + kh * 16 + 8);
      v16bf a1 = ldfrag(ap1 + kh * 16, ap1 + kh * 16 + 8);
      v16bf bq[4];
      #pragma unroll
      for (int t = 0; t < 4; ++t) {
        const bf16* bp = &Bsmem[buf][(wn * 64 + t * 16 + r) * 64 + kc * 32 + kh * 16];
        bq[t] = ldfrag(bp, bp + 8);
      }
      #pragma unroll
      for (int t = 0; t < 4; ++t) {
        acc[0][t] = wmma_bf16(a0, bq[t], acc[0][t]);
        acc[1][t] = wmma_bf16(a1, bq[t], acc[1][t]);
      }
    }
  };

  // ---- double-buffered main loop ----
  issue_stage(0, 0);
  wait_asynccnt0();
  __syncthreads();
  int buf = 0;
  for (int k0 = 0; k0 < K; k0 += 64) {
    int nk = k0 + 64;
    if (nk < K) issue_stage(nk, buf ^ 1);
    compute_stage(buf);
    if (nk < K) {
      wait_asynccnt0();      // our next-stage copies landed in LDS
      __syncthreads();       // everyone's copies landed / reads of cur done
    }
    buf ^= 1;
  }

  // ---- epilogue ----
  #pragma unroll
  for (int mi = 0; mi < 2; ++mi) {
    #pragma unroll
    for (int t = 0; t < 4; ++t) {
      #pragma unroll
      for (int v = 0; v < 8; ++v) {
        size_t off = (size_t)(mBase + wm * 32 + mi * 16 + v + 8 * kh) * N
                   + nBase + wn * 64 + t * 16 + r;
        if (Cf) Cf[off] = acc[mi][t][v];
        else    Cb[off] = (bf16)acc[mi][t][v];
      }
    }
  }
}

// ---------------------------------------------------------------------------
// Stage 2: RoPE on Q/K, scatter to head-major layouts (V transposed to [d][s])
// ---------------------------------------------------------------------------
__global__ void rope_scatter(const bf16* __restrict__ qkv,
                             const float* __restrict__ cosT, const float* __restrict__ sinT,
                             bf16* __restrict__ qO, bf16* __restrict__ kO, bf16* __restrict__ vT) {
  const int total = CB * CS * CQKV;
  int idx = blockIdx.x * blockDim.x + threadIdx.x;
  int stride = gridDim.x * blockDim.x;
  for (; idx < total; idx += stride) {
    int c  = idx % CQKV;
    int bs = idx / CQKV;
    int s  = bs % CS;
    int b  = bs / CS;
    int d    = c % CD;
    int head = c / CD;

    float val = (float)qkv[idx];
    if (head < CH + CKV) {
      // RoPE: rotate_half with half = 64
      int pidx = (d < 64) ? idx + 64 : idx - 64;
      float partner = (float)qkv[pidx];
      float rot = (d < 64) ? -partner : partner;
      size_t cs = (size_t)(b * CS + s) * CD + d;
      val = val * cosT[cs] + rot * sinT[cs];
      if (head < CH) {
        qO[((size_t)(b * CH + head) * CS + s) * CD + d] = (bf16)val;
      } else {
        int j = head - CH;
        kO[((size_t)(b * CKV + j) * CS + s) * CD + d] = (bf16)val;
      }
    } else {
      int j = head - CH - CKV;
      vT[((size_t)(b * CKV + j) * CD + d) * CS + s] = (bf16)val;   // transposed store
    }
  }
}

// ---------------------------------------------------------------------------
// Stage 3: causal GQA flash attention.
// One wave32 per (b, h, 16-row q tile); 32 keys per iteration.
// ---------------------------------------------------------------------------
__global__ void __launch_bounds__(32)
flash_attn(const bf16* __restrict__ Q, const bf16* __restrict__ Kk,
           const bf16* __restrict__ VT, bf16* __restrict__ O) {
  __shared__ bf16 ldsP[16 * 32];             // P staging: C-layout -> A-layout

  const int lane = threadIdx.x & 31;
  const int r    = lane & 15;
  const int kh   = lane >> 4;

  const int wid   = blockIdx.x;
  const int qb    = wid & (CS / 16 - 1);          // 128 q tiles
  const int h     = (wid >> 7) & (CH - 1);        // 32 heads
  const int b     = wid >> 12;                    // batch
  const int kvh   = h / CG;
  const int qbase = qb * 16;
  const float scale = 0.08838834764831845f;       // 1/sqrt(128)

  // Q fragments: 16 rows x 128 (K-dim) = 4 A-fragments, kept in registers
  v16bf qf[4];
  const bf16* qrow = Q + ((size_t)(b * CH + h) * CS + qbase + r) * CD;
  #pragma unroll
  for (int kc = 0; kc < 4; ++kc)
    qf[kc] = ldfrag(qrow + kc * 32 + kh * 16, qrow + kc * 32 + kh * 16 + 8);

  v8f o[8];
  #pragma unroll
  for (int dt = 0; dt < 8; ++dt) o[dt] = (v8f)(0.0f);
  float m_i[8], l_i[8];
  #pragma unroll
  for (int v = 0; v < 8; ++v) { m_i[v] = -__builtin_inff(); l_i[v] = 0.0f; }

  const bf16* kbase = Kk + (size_t)(b * CKV + kvh) * CS * CD;
  const bf16* vbase = VT + (size_t)(b * CKV + kvh) * CD * CS;

  for (int s0 = 0; s0 < qbase + 16; s0 += 32) {
    // --- scores: two 16x16 tiles over 128-deep K ---
    v8f sc0 = (v8f)(0.0f), sc1 = (v8f)(0.0f);
    const bf16* kr0 = kbase + (size_t)(s0 + r) * CD;
    const bf16* kr1 = kbase + (size_t)(s0 + 16 + r) * CD;
    #pragma unroll
    for (int kc = 0; kc < 4; ++kc) {
      v16bf b0 = ldfrag(kr0 + kc * 32 + kh * 16, kr0 + kc * 32 + kh * 16 + 8);
      sc0 = wmma_bf16(qf[kc], b0, sc0);
      v16bf b1 = ldfrag(kr1 + kc * 32 + kh * 16, kr1 + kc * 32 + kh * 16 + 8);
      sc1 = wmma_bf16(qf[kc], b1, sc1);
    }

    // --- scale + causal mask + online softmax ---
    float p0[8], p1[8], alpha[8];
    #pragma unroll
    for (int v = 0; v < 8; ++v) {
      int row = qbase + v + 8 * kh;
      int c0 = s0 + r, c1 = s0 + 16 + r;
      float a0 = (c0 <= row) ? sc0[v] * scale : -__builtin_inff();
      float a1 = (c1 <= row) ? sc1[v] * scale : -__builtin_inff();
      // row max across the 16 lanes of this half
      float t = fmaxf(a0, a1);
      t = fmaxf(t, __shfl_xor(t, 1, 16));
      t = fmaxf(t, __shfl_xor(t, 2, 16));
      t = fmaxf(t, __shfl_xor(t, 4, 16));
      t = fmaxf(t, __shfl_xor(t, 8, 16));
      float m_new = fmaxf(m_i[v], t);
      alpha[v] = __expf(m_i[v] - m_new);
      m_i[v] = m_new;
      p0[v] = __expf(a0 - m_new);
      p1[v] = __expf(a1 - m_new);
      float rs = p0[v] + p1[v];
      rs += __shfl_xor(rs, 1, 16);
      rs += __shfl_xor(rs, 2, 16);
      rs += __shfl_xor(rs, 4, 16);
      rs += __shfl_xor(rs, 8, 16);
      l_i[v] = l_i[v] * alpha[v] + rs;
    }
    // rescale existing accumulators
    #pragma unroll
    for (int dt = 0; dt < 8; ++dt)
      #pragma unroll
      for (int v = 0; v < 8; ++v) o[dt][v] *= alpha[v];

    // --- stage P (16x32 bf16) through LDS: C-layout -> A-layout ---
    #pragma unroll
    for (int v = 0; v < 8; ++v) {
      int row = v + 8 * kh;
      ldsP[row * 32 + r]      = (bf16)p0[v];
      ldsP[row * 32 + 16 + r] = (bf16)p1[v];
    }
    __syncthreads();   // single-wave WG: DS-count wait
    v16bf pf = ldfrag(&ldsP[r * 32 + kh * 16], &ldsP[r * 32 + kh * 16 + 8]);
    __syncthreads();

    // --- P (16x32) @ V (32x128): 8 output tiles, V^T gives contiguous frags ---
    #pragma unroll
    for (int dt = 0; dt < 8; ++dt) {
      const bf16* vp = vbase + (size_t)(dt * 16 + r) * CS + s0 + kh * 16;
      v16bf vf = ldfrag(vp, vp + 8);
      o[dt] = wmma_bf16(pf, vf, o[dt]);
    }
  }

  // --- normalize and store attn output, token-major [b, s, h*D + d] bf16 ---
  #pragma unroll
  for (int v = 0; v < 8; ++v) {
    float inv = 1.0f / l_i[v];
    int row = qbase + v + 8 * kh;
    size_t base = ((size_t)b * CS + row) * (CH * CD) + (size_t)h * CD;
    #pragma unroll
    for (int dt = 0; dt < 8; ++dt)
      O[base + dt * 16 + r] = (bf16)(o[dt][v] * inv);
  }
}

// ---------------------------------------------------------------------------
// Host launcher
// ---------------------------------------------------------------------------
extern "C" void kernel_launch(void* const* d_in, const int* in_sizes, int n_in,
                              void* d_out, int out_size, void* d_ws, size_t ws_size,
                              hipStream_t stream) {
  const float* hidden = (const float*)d_in[0];   // [B,S,HID]
  const float* w_qkv  = (const float*)d_in[1];   // [(H+2KV)*D, HID]
  const float* w_o    = (const float*)d_in[2];   // [HID, H*D]
  const float* cosT   = (const float*)d_in[3];   // [B,S,D]
  const float* sinT   = (const float*)d_in[4];   // [B,S,D]
  float* out = (float*)d_out;                    // [B,S,HID]

  const int nHid  = CB * CS * CHID;      // 16,777,216
  const int nWqkv = CQKV * CHID;         // 25,165,824
  const int nWo   = CHID * CH * CD;      // 16,777,216
  const int nQkv  = CB * CS * CQKV;      // 25,165,824
  const int nQ    = CB * CH * CS * CD;   // 16,777,216
  const int nK    = CB * CKV * CS * CD;  //  4,194,304
  const int nAttn = CB * CS * CH * CD;   // 16,777,216

  size_t off = 0;
  auto carve = [&](size_t elems) {
    void* p = (char*)d_ws + off;
    off += (elems * sizeof(bf16) + 255) & ~(size_t)255;
    return (bf16*)p;
  };
  bf16* hid_bf  = carve(nHid);
  bf16* wqkv_bf = carve(nWqkv);
  bf16* wo_bf   = carve(nWo);
  bf16* qkv_bf  = carve(nQkv);
  bf16* q_bf    = carve(nQ);
  bf16* k_bf    = carve(nK);
  bf16* vT_bf   = carve(nK);
  bf16* attn_bf = carve(nAttn);
  (void)ws_size; (void)in_sizes; (void)n_in; (void)out_size;

  // Stage 0: conversions
  cvt_f32_bf16<<<(nHid  + 255) / 256, 256, 0, stream>>>(hidden, hid_bf,  nHid);
  cvt_f32_bf16<<<(nWqkv + 255) / 256, 256, 0, stream>>>(w_qkv,  wqkv_bf, nWqkv);
  cvt_f32_bf16<<<(nWo   + 255) / 256, 256, 0, stream>>>(w_o,    wo_bf,   nWo);

  // Stage 1: QKV = hidden @ w_qkv^T   (M=4096, N=6144, K=4096)
  {
    int M = CB * CS, N = CQKV, K = CHID;
    int blocks = (M / 128) * (N / 128);        // 1536
    wmma_gemm_nt_lds<<<blocks, 256, 0, stream>>>(hid_bf, wqkv_bf, nullptr, qkv_bf, M, N, K);
  }

  // Stage 2: RoPE + scatter
  rope_scatter<<<(nQkv + 255) / 256, 256, 0, stream>>>(qkv_bf, cosT, sinT, q_bf, k_bf, vT_bf);

  // Stage 3: flash attention — one wave per (b, h, q-tile)
  flash_attn<<<CB * CH * (CS / 16), 32, 0, stream>>>(q_bf, k_bf, vT_bf, attn_bf);

  // Stage 4: out = attn @ w_o^T   (M=4096, N=4096, K=4096) -> fp32
  {
    int M = CB * CS, N = CHID, K = CH * CD;
    int blocks = (M / 128) * (N / 128);        // 1024
    wmma_gemm_nt_lds<<<blocks, 256, 0, stream>>>(attn_bf, wo_bf, out, nullptr, M, N, K);
  }
}
